// MatMul_412316860460
// MI455X (gfx1250) — 8-bit matrix operand order repaired and checked by reading, compiled, not run on hardware
//
#include <hip/hip_runtime.h>
#include <stdint.h>


typedef __attribute__((ext_vector_type(8))) int v8i;

static constexpr int BB = 8;
static constexpr int MM = 4096;
static constexpr int KK = 64;
static constexpr int NN = 4096;

__device__ __forceinline__ uint32_t quant1(float x, float c, float qs) {
    float xc = fminf(fmaxf(x, 0.0f), c);
    return (uint32_t)(int)rintf(xc * qs);
}

__global__ void __launch_bounds__(256)
pack_a(const float* __restrict__ x1, uint8_t* __restrict__ ap,
       const float* __restrict__ clip, const int* __restrict__ level) {
    int r = blockIdx.x * blockDim.x + threadIdx.x;
    if (r >= BB * MM) return;
    const float c  = clip[0];
    const float qs = (float)(level[0] - 1) / c;

    const float* src = x1 + (size_t)r * KK;
    uint32_t w[16];
#pragma unroll
    for (int j = 0; j < 16; ++j) {
        int half  = j >> 3;
        int chunk = (j & 7) >> 1;
        int q     = j & 1;
        int k0    = chunk * 16 + half * 8 + q * 4;
        float4 t  = *(const float4*)(src + k0);
        w[j] = quant1(t.x, c, qs) | (quant1(t.y, c, qs) << 8) |
               (quant1(t.z, c, qs) << 16) | (quant1(t.w, c, qs) << 24);
    }
    uint4* dst = (uint4*)(ap + (size_t)r * 64);
    dst[0] = make_uint4(w[0],  w[1],  w[2],  w[3]);
    dst[1] = make_uint4(w[4],  w[5],  w[6],  w[7]);
    dst[2] = make_uint4(w[8],  w[9],  w[10], w[11]);
    dst[3] = make_uint4(w[12], w[13], w[14], w[15]);
}

__global__ void __launch_bounds__(256)
pack_b(const float* __restrict__ x2, uint8_t* __restrict__ bp,
       const float* __restrict__ clip, const int* __restrict__ level) {
    int n = blockIdx.x * blockDim.x + threadIdx.x;
    int b = blockIdx.y;
    const float c  = clip[0];
    const float qs = (float)(level[0] - 1) / c;

    const float* src = x2 + (size_t)b * KK * NN + n;
    uint32_t w[16];
#pragma unroll
    for (int j = 0; j < 16; ++j) {
        uint32_t v = 0;
#pragma unroll
        for (int by = 0; by < 4; ++by) {
            int idx = j * 4 + by;
            int h   = idx >> 5;
            int jj  = idx & 31;
            int g   = jj >> 4;
            int kk  = jj & 15;
            int k   = h * 8 + (jj >> 3) * 16 + (jj & 7);
            v |= quant1(src[(size_t)k * NN], c, qs) << (8 * by);
        }
        w[j] = v;
    }
    uint4* dst = (uint4*)(bp + ((size_t)b * NN + n) * 64);
    dst[0] = make_uint4(w[0],  w[1],  w[2],  w[3]);
    dst[1] = make_uint4(w[4],  w[5],  w[6],  w[7]);
    dst[2] = make_uint4(w[8],  w[9],  w[10], w[11]);
    dst[3] = make_uint4(w[12], w[13], w[14], w[15]);
}

__global__ void __launch_bounds__(256)
gemm_iu8(const uint8_t* __restrict__ ap, const uint8_t* __restrict__ bp,
         float* __restrict__ out,
         const float* __restrict__ c1p, const float* __restrict__ c2p,
         const int* __restrict__ level) {
    __shared__ alignas(16) uint8_t lbs[8192];

    const int n_blk = blockIdx.x;
    const int m_blk = blockIdx.y;
    const int b     = blockIdx.z;
    const int tid   = threadIdx.x;
    const int wave  = tid >> 5;
    const int lane  = tid & 31;
    const int lrow  = lane & 15;
    const int lhalf = lane >> 4;

    {
        const uint8_t* gsrc =
            bp + ((size_t)b * NN + n_blk * 128) * 64 + (size_t)tid * 32;
        uint32_t loff = (uint32_t)(uintptr_t)(&lbs[tid * 32]);
        asm volatile("global_load_async_to_lds_b128 %0, %1, off"
                     :: "v"(loff), "v"(gsrc) : "memory");
        asm volatile("global_load_async_to_lds_b128 %0, %1, off offset:16"
                     :: "v"(loff), "v"(gsrc) : "memory");
    }

    const int m0 = m_blk * 256 + wave * 32;
    const v8i a0 =
        *(const v8i*)(ap + ((size_t)(b * MM + m0 + lrow)) * 64 + lhalf * 32);
    const v8i a1 =
        *(const v8i*)(ap + ((size_t)(b * MM + m0 + 16 + lrow)) * 64 + lhalf * 32);

    const float Lm1   = (float)(level[0] - 1);
    const float scale = c1p[0] * c2p[0] * __builtin_amdgcn_rcpf(Lm1 * Lm1);

    asm volatile("s_wait_asynccnt 0" ::: "memory");
    __syncthreads();

    const int    n_base = n_blk * 128;
    const size_t row0   = ((size_t)(b * MM + m0 + lhalf * 8)) * NN;
    const size_t row1   = row0 + (size_t)16 * NN;

#pragma unroll
    for (int t = 0; t < 8; ++t) {
        const v8i bf = *(const v8i*)(&lbs[(t * 16 + lrow) * 64 + lhalf * 32]);

        v8i acc0 = {};
        v8i acc1 = {};
        acc0 = __builtin_amdgcn_wmma_i32_16x16x64_iu8(
            false, a0, false, bf, acc0, false, false);
        acc1 = __builtin_amdgcn_wmma_i32_16x16x64_iu8(
            false, a1, false, bf, acc1, false, false);

        const int n0 = n_base + t * 16;
        float* o0 = out + row0 + n0 + lrow;
        float* o1 = out + row1 + n0 + lrow;
#pragma unroll
        for (int v = 0; v < 8; ++v) {
            o0[(size_t)v * NN] = (float)acc0[v] * scale;
        }
#pragma unroll
        for (int v = 0; v < 8; ++v) {
            o1[(size_t)v * NN] = (float)acc1[v] * scale;
        }
    }
}

extern "C" void kernel_launch(void* const* d_in, const int* in_sizes, int n_in,
                              void* d_out, int out_size, void* d_ws, size_t ws_size,
                              hipStream_t stream) {
    (void)in_sizes; (void)n_in; (void)out_size; (void)ws_size;
    const float* x1  = (const float*)d_in[0];
    const float* x2  = (const float*)d_in[1];
    const float* c1  = (const float*)d_in[2];
    const float* c2  = (const float*)d_in[3];
    const int*   lvl = (const int*)d_in[4];
    float* out = (float*)d_out;

    uint8_t* ap = (uint8_t*)d_ws;
    uint8_t* bp = ap + (size_t)BB * MM * KK;

    pack_a<<<(BB * MM + 255) / 256, 256, 0, stream>>>(x1, ap, c1, lvl);
    pack_b<<<dim3(NN / 256, BB), 256, 0, stream>>>(x2, bp, c2, lvl);
    gemm_iu8<<<dim3(NN / 128, MM / 256, BB), 256, 0, stream>>>(
        ap, bp, out, c1, c2, lvl);
}
